// Encoder_48430051229941
// MI455X (gfx1250) — matrix operand order repaired and checked by reading, not compiled, not run on hardware
//
#include <hip/hip_runtime.h>
#include <stdint.h>

// ---------------------------------------------------------------------------
// Sigma-delta conv encoder, fully fused per layer on gfx1250 (wave32 WMMA).
//   conv(implicit GEMM, bf16 v_wmma_f32_16x16x32_bf16, f32 accum)
//   + cumsum over T + relu + delta encode : all in registers, per wave.
// ---------------------------------------------------------------------------

typedef __attribute__((ext_vector_type(16))) __bf16 v16bf;
typedef __attribute__((ext_vector_type(8)))  __bf16 v8bf;
typedef __attribute__((ext_vector_type(8)))  float  v8f;

#define N_BATCH 8
#define T_STEPS 16
#define NT      128           // N_BATCH * T_STEPS independent conv images

// Layer geometry
//  L1 (via im2col): C=32(pad from 18), 1x1/stride1 on 63x63, O=64,  P=3969
//  L2: C=64,  3x3 stride2, 63->31, O=128, P=961
//  L3: C=128, 3x3 stride2, 31->15, O=256, P=225
static constexpr size_t align_up(size_t x, size_t a) { return (x + a - 1) / a * a; }

static constexpr size_t OFF_SUMS = 0;                                    // 3 floats
static constexpr size_t OFF_WA1  = 256;                                  // 64 x 32  bf16
static constexpr size_t OFF_WA2  = align_up(OFF_WA1 + 64ull  * 32   * 2, 256);  // 128 x 576
static constexpr size_t OFF_WA3  = align_up(OFF_WA2 + 128ull * 576  * 2, 256);  // 256 x 1152
static constexpr size_t OFF_R1   = align_up(OFF_WA3 + 256ull * 1152 * 2, 1024); // im2col(L1) then x2
static constexpr size_t SZ_R1    = 128ull * 3969 * 32 * 2;               // 32.5MB (>= x2: 31.5MB)
static constexpr size_t OFF_R2   = align_up(OFF_R1 + SZ_R1, 1024);       // x1 bf16: 65MB

static constexpr size_t OUT_X_ELEMS = 8ull * 256 * 225 * 16;             // 7,372,800

// ---------------------------------------------------------------------------
__global__ void k_init(float* sums) {
    if (threadIdx.x < 3) sums[threadIdx.x] = 0.f;
}

// Weight normalization -> bf16 GEMM A matrix, k = (kh*3+kw)*Cin + c, zero-pad to Kpad.
__global__ void k_wnorm(const float* __restrict__ v, const float* __restrict__ g,
                        __bf16* __restrict__ A, int O, int Cin, int Kpad) {
    int o = blockIdx.x * blockDim.x + threadIdx.x;
    if (o >= O) return;
    const float* vo = v + (size_t)o * Cin * 9;   // v layout (O, Cin, 3, 3)
    float ss = 0.f;
    for (int i = 0; i < Cin * 9; ++i) { float t = vo[i]; ss += t * t; }
    float scale = g[o] / sqrtf(ss);
    __bf16* Ao = A + (size_t)o * Kpad;
    for (int kk = 0; kk < 9; ++kk)
        for (int c = 0; c < Cin; ++c)
            Ao[kk * Cin + c] = (__bf16)(scale * vo[(size_t)c * 9 + kk]);
    for (int k = 9 * Cin; k < Kpad; ++k) Ao[k] = (__bf16)0.f;
}

// Layer-1 im2col: spike (8,2,128,128,16) f32 -> (img=n*16+t, 63, 63, 32) bf16, K padded 18->32
__global__ void k_im2col1(const float* __restrict__ spike, __bf16* __restrict__ Xc) {
    const int P = 63 * 63;
    int idx = blockIdx.x * blockDim.x + threadIdx.x;    // over 128 * 3969
    if (idx >= NT * P) return;
    int img = idx / P, p = idx - img * P;
    int n = img >> 4, t = img & 15;
    int ho = p / 63, wo = p - ho * 63;
    __bf16* out = Xc + (size_t)idx * 32;
    #pragma unroll
    for (int kk = 0; kk < 9; ++kk) {
        int kh = kk / 3, kw = kk - kh * 3;
        int h = ho * 2 + kh, w = wo * 2 + kw;
        #pragma unroll
        for (int c = 0; c < 2; ++c) {
            float val = spike[((((size_t)n * 2 + c) * 128 + h) * 128 + w) * 16 + t];
            out[kk * 2 + c] = (__bf16)val;
        }
    }
    #pragma unroll
    for (int k = 18; k < 32; ++k) out[k] = (__bf16)0.f;
}

// ---------------------------------------------------------------------------
// Fused layer: one wave computes a 16(O) x 16(P) tile for ALL 16 time steps.
// K loop: load A fragment once, issue 16 WMMA (one per t, B from image n*16+t).
// Then the sigma/relu/delta recurrence runs entirely in registers.
// ---------------------------------------------------------------------------
template<int C, int KH, int KW, int STRIDE, int Ho, int Wo, int O, bool FINAL>
__global__ __launch_bounds__(32)
void k_fused_layer(const __bf16* __restrict__ A,   // (O, K) bf16, k=(kh*KW+kw)*C+c
                   const __bf16* __restrict__ X,   // (img, Win, Win, C) bf16
                   __bf16* __restrict__ Snext,     // (img, P, O) bf16   (FINAL=false)
                   float*  __restrict__ Sout,      // (N, O, P, T) f32   (FINAL=true)
                   float*  __restrict__ sumAcc) {
    constexpr int P       = Ho * Wo;
    constexpr int Win     = (Wo - 1) * STRIDE + KW;
    constexpr int K       = KH * KW * C;
    constexpr int KSTEPS  = K / 32;
    constexpr int CSTEPS  = C / 32;
    constexpr size_t IMGSZ = (size_t)Win * Win * C;

    const int lane = threadIdx.x;
    const int ncol = lane & 15;       // GEMM N (spatial) / A row index
    const int hi   = lane >> 4;       // lane half selects K sub-slices & D rows
    const int n    = blockIdx.z;      // batch
    const int m0   = blockIdx.y * 16; // output-channel tile base

    int p = blockIdx.x * 16 + ncol;
    const bool valid = (p < P);
    if (!valid) p = P - 1;            // clamp: EXEC stays all-ones for WMMA
    const int ho = p / Wo;
    const int wo = p - ho * Wo;

    const __bf16* xb = X + (size_t)n * T_STEPS * IMGSZ
                         + ((size_t)(ho * STRIDE) * Win + (size_t)(wo * STRIDE)) * C;
    const __bf16* arow = A + (size_t)(m0 + ncol) * K + hi * 8;

    const v8f zero8 = {0.f,0.f,0.f,0.f,0.f,0.f,0.f,0.f};
    v8f acc[T_STEPS];
    #pragma unroll
    for (int t = 0; t < T_STEPS; ++t) acc[t] = zero8;

    for (int s = 0; s < KSTEPS; ++s) {
        // A fragment (ISA 16-bit 16x32 layout): lane half hi -> K {0-7,16-23}/{8-15,24-31}
        const __bf16* pa = arow + s * 32;
        v8bf alo = *reinterpret_cast<const v8bf*>(pa);
        v8bf ahi = *reinterpret_cast<const v8bf*>(pa + 16);
        v16bf afrag;
        #pragma unroll
        for (int i = 0; i < 8; ++i) { afrag[i] = alo[i]; afrag[i + 8] = ahi[i]; }

        // Decompose K step -> (kh, kw, c0); activation slice is contiguous in C
        const int kk = s / CSTEPS;
        const int c0 = (s - kk * CSTEPS) * 32;
        const int kh = kk / KW;
        const int kw = kk - kh * KW;
        const __bf16* pb0 = xb + ((size_t)kh * Win + kw) * C + c0 + hi * 8;

        #pragma unroll
        for (int t = 0; t < T_STEPS; ++t) {
            // B fragment (32x16): lane half hi holds K {0-7,16-23}/{8-15,24-31}
            const __bf16* pb = pb0 + (size_t)t * IMGSZ;
            v8bf blo = *reinterpret_cast<const v8bf*>(pb);
            v8bf bhi = *reinterpret_cast<const v8bf*>(pb + 16);
            v16bf bfrag;
            #pragma unroll
            for (int i = 0; i < 8; ++i) { bfrag[i] = blo[i]; bfrag[i + 8] = bhi[i]; }
            acc[t] = __builtin_amdgcn_wmma_f32_16x16x32_bf16(
                false, afrag, false, bfrag, (short)0, acc[t], false, false);
        }
    }

    // In-register temporal chain: sigma (cumsum) -> relu -> delta encode.
    // D-fragment row r in this lane is output channel o = m0 + 8*hi + r, column = p.
    float sig[8], prev[8], res[8];
    #pragma unroll
    for (int r = 0; r < 8; ++r) { sig[r] = 0.f; prev[r] = 0.f; res[r] = 0.f; }
    float lsum = 0.f;

    #pragma unroll
    for (int t = 0; t < T_STEPS; ++t) {
        v8bf pack;
        #pragma unroll
        for (int r = 0; r < 8; ++r) {
            sig[r] += acc[t][r];
            float a = fmaxf(sig[r], 0.f);
            float d = a - prev[r] + res[r];
            float sp = (fabsf(d) >= 0.1f) ? d : 0.f;
            res[r]  = d - sp;
            prev[r] = a;
            if (valid) lsum += sp;
            if (FINAL) {
                if (valid) {
                    int o = m0 + hi * 8 + r;
                    Sout[(((size_t)n * O + o) * P + p) * T_STEPS + t] = sp;
                }
            } else {
                pack[r] = (__bf16)sp;
            }
        }
        if (!FINAL && valid) {
            __bf16* dst = Snext + (((size_t)n * T_STEPS + t) * P + p) * O + m0 + hi * 8;
            *reinterpret_cast<v8bf*>(dst) = pack;   // 16B NHWC store feeds next layer
        }
    }

    // wave32 reduction, one atomic per wave (layer spike-count mean numerator)
    #pragma unroll
    for (int m = 16; m > 0; m >>= 1) lsum += __shfl_xor(lsum, m, 32);
    if (lane == 0) atomicAdd(sumAcc, lsum);
}

__global__ void k_finalize(const float* __restrict__ sums, float* __restrict__ cnt) {
    if (threadIdx.x == 0) {
        cnt[0] = sums[0] / 32514048.f;   // 8*64*63*63*16
        cnt[1] = sums[1] / 15745024.f;   // 8*128*31*31*16
        cnt[2] = sums[2] / 7372800.f;    // 8*256*15*15*16
    }
}

// ---------------------------------------------------------------------------
extern "C" void kernel_launch(void* const* d_in, const int* in_sizes, int n_in,
                              void* d_out, int out_size, void* d_ws, size_t ws_size,
                              hipStream_t stream) {
    (void)in_sizes; (void)n_in; (void)out_size; (void)ws_size;

    const float* spike = (const float*)d_in[0];
    const float* v1 = (const float*)d_in[1];
    const float* g1 = (const float*)d_in[2];
    const float* v2 = (const float*)d_in[3];
    const float* g2 = (const float*)d_in[4];
    const float* v3 = (const float*)d_in[5];
    const float* g3 = (const float*)d_in[6];
    float* out = (float*)d_out;

    char* ws = (char*)d_ws;
    float*  sums  = (float*)(ws + OFF_SUMS);
    __bf16* wA1   = (__bf16*)(ws + OFF_WA1);
    __bf16* wA2   = (__bf16*)(ws + OFF_WA2);
    __bf16* wA3   = (__bf16*)(ws + OFF_WA3);
    __bf16* bufR1 = (__bf16*)(ws + OFF_R1);   // im2col for L1, then x2
    __bf16* bufX1 = (__bf16*)(ws + OFF_R2);   // x1

    k_init<<<1, 32, 0, stream>>>(sums);
    k_wnorm<<<1, 64, 0, stream>>>(v1, g1, wA1,  64,   2,   32);
    k_wnorm<<<2, 64, 0, stream>>>(v2, g2, wA2, 128,  64,  576);
    k_wnorm<<<4, 64, 0, stream>>>(v3, g3, wA3, 256, 128, 1152);

    {
        int total = NT * 3969;
        k_im2col1<<<(total + 255) / 256, 256, 0, stream>>>(spike, bufR1);
    }

    // L1: 63x63 "1x1" conv on im2col buffer, K=32, O=64   -> x1 (bf16 NHWC)
    k_fused_layer<32, 1, 1, 1, 63, 63, 64, false>
        <<<dim3(249, 4, N_BATCH), 32, 0, stream>>>(wA1, bufR1, bufX1, nullptr, sums + 0);
    // L2: 3x3 s2, 63->31, K=576, O=128                    -> x2 (bf16, reuses R1)
    k_fused_layer<64, 3, 3, 2, 31, 31, 128, false>
        <<<dim3(61, 8, N_BATCH), 32, 0, stream>>>(wA2, bufX1, bufR1, nullptr, sums + 1);
    // L3: 3x3 s2, 31->15, K=1152, O=256                   -> d_out (f32, NCHW-T)
    k_fused_layer<128, 3, 3, 2, 15, 15, 256, true>
        <<<dim3(15, 16, N_BATCH), 32, 0, stream>>>(wA3, bufR1, nullptr, out, sums + 2);

    k_finalize<<<1, 32, 0, stream>>>(sums, out + OUT_X_ELEMS);
}
